// Mamba2LMHeadModel_22728966931071
// MI455X (gfx1250) — matrix operand order repaired and checked by reading, not compiled, not run on hardware
//
#include <hip/hip_runtime.h>
#include <hip/hip_bf16.h>
#include <math.h>

// ---------------- model dims ----------------
#define NLAYER    4
#define DMODEL    1024
#define DSTATE    128
#define DCONV     4
#define HEADDIM   64
#define CHUNKL    64
#define NCHUNK    16
#define SEQLEN    1024
#define DINNER    2048
#define NHEADS    32
#define DINPROJ   4384   // 2*DINNER + 2*DSTATE + NHEADS
#define CONVDIM   2304   // DINNER + 2*DSTATE
#define VOCAB     50288
#define EPSF      1e-5f

// ---------------- vector types (CDNA5 WMMA) ----------------
typedef __bf16 bf16_t;
typedef bf16_t v16bf __attribute__((ext_vector_type(16)));
typedef bf16_t v8bf  __attribute__((ext_vector_type(8)));
typedef bf16_t v4bf  __attribute__((ext_vector_type(4)));
typedef float  v8f   __attribute__((ext_vector_type(8)));
typedef float  v4f   __attribute__((ext_vector_type(4)));

// D = A*B + C, A:16x32 bf16, B:32x16 bf16, C/D:16x16 f32  -> v_wmma_f32_16x16x32_bf16
__device__ __forceinline__ v8f wmma_bf16(v16bf a, v16bf b, v8f c) {
  return __builtin_amdgcn_wmma_f32_16x16x32_bf16(false, a, false, b, (short)0, c,
                                                 false, false);
}

// A-fragment: row-major [m][k] tile in LDS, lane = row (m = lane&15),
// K runs: {hi*8 .. hi*8+7} and {16+hi*8 .. 16+hi*8+7}  (hi = lane>>4)
__device__ __forceinline__ v16bf frag_a(const bf16_t* tile, int stride, int lane, int kk) {
  const int row = lane & 15, hi = lane >> 4;
  const bf16_t* p = tile + row * stride + kk + hi * 8;
  union { v16bf v; v8bf h[2]; } u;
  u.h[0] = *(const v8bf*)(p);
  u.h[1] = *(const v8bf*)(p + 16);
  return u.v;
}

// B-fragment: operand stored row-major [n][k] (i.e. B^T), lane = column n,
// K runs: {hi*8 .. hi*8+7} and {16+hi*8 .. 16+hi*8+7}  (hi = lane>>4)
__device__ __forceinline__ v16bf frag_b(const bf16_t* tile, int stride, int lane, int kk) {
  const int col = lane & 15, hi = lane >> 4;
  const bf16_t* p = tile + col * stride + kk + hi * 8;
  union { v16bf v; v8bf h[2]; } u;
  u.h[0] = *(const v8bf*)(p);
  u.h[1] = *(const v8bf*)(p + 16);
  return u.v;
}

// ======================================================================
// GEMM:  C[M,N] = A[M,K] @ W[N,K]^T  (+ resid), fp32 in HBM, bf16 WMMA
// block tile 128x128, BK=64, 256 threads = 8 waves (4x2), wave tile 32x64
// ======================================================================
#define GBM 128
#define GBN 128
#define GBK 64
#define GKP 72   // padded LDS row (bf16 elems): 144B rows -> conflict-free b128

__global__ __launch_bounds__(256) void gemm_nt_bf16(
    const float* __restrict__ A, const float* __restrict__ Bw,
    float* __restrict__ C, const float* __restrict__ resid,
    int M, int N, int K)
{
  __shared__ bf16_t sA[GBM * GKP];
  __shared__ bf16_t sB[GBN * GKP];
  const int tid  = threadIdx.x;
  const int lane = tid & 31;
  const int wave = tid >> 5;
  const int wr = wave >> 1, wc = wave & 1;
  const int frow = lane & 15, fhi = lane >> 4;
  const int m0 = blockIdx.y * GBM;
  const int n0 = blockIdx.x * GBN;

  v8f acc[2][4] = {};

  const int lr = tid >> 4;         // 0..15 (row within pass)
  const int lc = (tid & 15) * 4;   // 0..60 (col, float4 granule)

  for (int k0 = 0; k0 < K; k0 += GBK) {
    __syncthreads();
#pragma unroll
    for (int p = 0; p < 8; ++p) {
      const int r = lr + p * 16;
      v4f av = *(const v4f*)(A + (size_t)(m0 + r) * K + (k0 + lc));
      v4bf ah; ah[0]=(bf16_t)av[0]; ah[1]=(bf16_t)av[1]; ah[2]=(bf16_t)av[2]; ah[3]=(bf16_t)av[3];
      *(v4bf*)(&sA[r * GKP + lc]) = ah;
      const int nr = n0 + r;
      v4f bv = {0.f, 0.f, 0.f, 0.f};
      if (nr < N) bv = *(const v4f*)(Bw + (size_t)nr * K + (k0 + lc));
      v4bf bh; bh[0]=(bf16_t)bv[0]; bh[1]=(bf16_t)bv[1]; bh[2]=(bf16_t)bv[2]; bh[3]=(bf16_t)bv[3];
      *(v4bf*)(&sB[r * GKP + lc]) = bh;
    }
    __syncthreads();
#pragma unroll
    for (int kk = 0; kk < GBK; kk += 32) {
      v16bf a0 = frag_a(&sA[(wr * 32 + 0 ) * GKP], GKP, lane, kk);
      v16bf a1 = frag_a(&sA[(wr * 32 + 16) * GKP], GKP, lane, kk);
      v16bf b0 = frag_b(&sB[(wc * 64 + 0 ) * GKP], GKP, lane, kk);
      v16bf b1 = frag_b(&sB[(wc * 64 + 16) * GKP], GKP, lane, kk);
      v16bf b2 = frag_b(&sB[(wc * 64 + 32) * GKP], GKP, lane, kk);
      v16bf b3 = frag_b(&sB[(wc * 64 + 48) * GKP], GKP, lane, kk);
      acc[0][0] = wmma_bf16(a0, b0, acc[0][0]);
      acc[0][1] = wmma_bf16(a0, b1, acc[0][1]);
      acc[0][2] = wmma_bf16(a0, b2, acc[0][2]);
      acc[0][3] = wmma_bf16(a0, b3, acc[0][3]);
      acc[1][0] = wmma_bf16(a1, b0, acc[1][0]);
      acc[1][1] = wmma_bf16(a1, b1, acc[1][1]);
      acc[1][2] = wmma_bf16(a1, b2, acc[1][2]);
      acc[1][3] = wmma_bf16(a1, b3, acc[1][3]);
    }
  }
  // epilogue: C/D layout -> lane holds col (lane&15), rows v + 8*(lane>>4)
#pragma unroll
  for (int ni = 0; ni < 4; ++ni) {
    const int n = n0 + wc * 64 + ni * 16 + frow;
    if (n >= N) continue;
#pragma unroll
    for (int mi = 0; mi < 2; ++mi) {
#pragma unroll
      for (int v = 0; v < 8; ++v) {
        const int m = m0 + wr * 32 + mi * 16 + v + 8 * fhi;
        const size_t o = (size_t)m * N + n;
        float val = acc[mi][ni][v];
        if (resid) val += resid[o];
        C[o] = val;
      }
    }
  }
}

// ======================================================================
// elementwise kernels
// ======================================================================
__global__ void embed_k(const int* __restrict__ ids, const float* __restrict__ emb,
                        float* __restrict__ x) {
  const int idx = blockIdx.x * 256 + threadIdx.x;
  if (idx >= SEQLEN * DMODEL) return;
  const int l = idx >> 10, d = idx & 1023;
  x[idx] = emb[(size_t)ids[l] * DMODEL + d];
}

__global__ __launch_bounds__(256) void rmsnorm_k(const float* __restrict__ x,
                                                 const float* __restrict__ w,
                                                 float* __restrict__ out) {
  __shared__ float red[256];
  const int l = blockIdx.x, tid = threadIdx.x;
  float vals[4]; float ss = 0.f;
#pragma unroll
  for (int e = 0; e < 4; ++e) {
    const int i = tid + e * 256;
    const float t = x[(size_t)l * DMODEL + i];
    vals[e] = t; ss += t * t;
  }
  red[tid] = ss; __syncthreads();
  for (int s = 128; s > 0; s >>= 1) { if (tid < s) red[tid] += red[tid + s]; __syncthreads(); }
  const float scale = rsqrtf(red[0] / (float)DMODEL + EPSF);
#pragma unroll
  for (int e = 0; e < 4; ++e) {
    const int i = tid + e * 256;
    out[(size_t)l * DMODEL + i] = vals[e] * scale * w[i];
  }
}

__global__ void conv_silu_k(const float* __restrict__ zx, const float* __restrict__ cw,
                            const float* __restrict__ cb, float* __restrict__ xconv) {
  const int idx = blockIdx.x * 256 + threadIdx.x;
  if (idx >= SEQLEN * CONVDIM) return;
  const int l = idx / CONVDIM, ch = idx - l * CONVDIM;
  float s = cb[ch];
#pragma unroll
  for (int j = 0; j < DCONV; ++j) {
    const int li = l + j - (DCONV - 1);
    if (li >= 0) s += zx[(size_t)li * DINPROJ + DINNER + ch] * cw[ch * DCONV + j];
  }
  xconv[idx] = s / (1.f + __expf(-s));   // silu
}

__global__ void dt_k(const float* __restrict__ zx, const float* __restrict__ dtb,
                     const float* __restrict__ A_log, float* __restrict__ dt_sp,
                     float* __restrict__ dA) {
  const int idx = blockIdx.x * 256 + threadIdx.x;
  if (idx >= SEQLEN * NHEADS) return;
  const int l = idx >> 5, h = idx & 31;
  const float x = zx[(size_t)l * DINPROJ + DINNER + CONVDIM + h] + dtb[h];
  const float sp = (x > 20.f) ? x : log1pf(__expf(x));
  dt_sp[idx] = sp;
  dA[idx] = -__expf(A_log[h]) * sp;
}

__global__ __launch_bounds__(256) void gated_rmsnorm_k(const float* __restrict__ Y,
                                                       const float* __restrict__ zx,
                                                       const float* __restrict__ gw,
                                                       float* __restrict__ out) {
  __shared__ float red[256];
  const int l = blockIdx.x, tid = threadIdx.x;
  float vals[8]; float ss = 0.f;
#pragma unroll
  for (int e = 0; e < 8; ++e) {
    const int i = tid + e * 256;
    const float z = zx[(size_t)l * DINPROJ + i];            // gate z
    const float g = z / (1.f + __expf(-z));                 // z*sigmoid(z)
    const float t = Y[(size_t)l * DINNER + i] * g;
    vals[e] = t; ss += t * t;
  }
  red[tid] = ss; __syncthreads();
  for (int s = 128; s > 0; s >>= 1) { if (tid < s) red[tid] += red[tid + s]; __syncthreads(); }
  const float scale = rsqrtf(red[0] / (float)DINNER + EPSF);
#pragma unroll
  for (int e = 0; e < 8; ++e) {
    const int i = tid + e * 256;
    out[(size_t)l * DINNER + i] = vals[e] * scale * gw[i];
  }
}

// ======================================================================
// SSD intra-chunk: per (chunk c, head h) block.
//   scores = C @ B^T          (WMMA, K=128)
//   W      = mask(scores) * exp(cs[l]-cs[s])
//   Ydiag  = W @ xdt          (WMMA, K=64)
//   states = (xdt*decay)^T @ B (WMMA, K=64)  -> [p=64, n=128]
// ======================================================================
#define SBST 136   // padded row for [*,128] bf16 tiles
#define SXT  72    // padded row for [*,64]  bf16 tiles

__global__ __launch_bounds__(256) void ssd_intra_k(
    const float* __restrict__ xconv,   // [L, CONVDIM]
    const float* __restrict__ dt_sp,   // [L, NHEADS]
    const float* __restrict__ dA,      // [L, NHEADS]
    float* __restrict__ a_cs,          // [NCHUNK, NHEADS, CHUNKL]
    float* __restrict__ Y,             // [L, DINNER]   (Y_diag out)
    float* __restrict__ states)        // [NCHUNK, NHEADS, HEADDIM, DSTATE]
{
  __shared__ bf16_t sC [64 * SBST];    // C chunk  [l][n]       (A-op)
  __shared__ bf16_t sBW[64 * SBST];    // B chunk  [s][n] (B-op), reused as W [l][s]
  __shared__ bf16_t sxdT[64 * SXT];    // (x*dt)^T [p][l]
  __shared__ bf16_t sBdT[128 * SXT];   // (B*decay)^T [n][l]
  __shared__ float s_cs[64], s_decay[64], s_dt[64];

  const int tid = threadIdx.x, lane = tid & 31, wave = tid >> 5;
  const int frow = lane & 15, fhi = lane >> 4;
  const int c = blockIdx.x, h = blockIdx.y;
  const int l0 = c * CHUNKL;

  if (tid < 64) {
    s_dt[tid] = dt_sp[(size_t)(l0 + tid) * NHEADS + h];
    s_cs[tid] = dA[(size_t)(l0 + tid) * NHEADS + h];
  }
  __syncthreads();
  // inclusive cumsum of dA within the chunk (Hillis-Steele)
  for (int off = 1; off < 64; off <<= 1) {
    float v = 0.f;
    if (tid < 64 && tid >= off) v = s_cs[tid - off];
    __syncthreads();
    if (tid < 64) s_cs[tid] += v;
    __syncthreads();
  }
  if (tid < 64) {
    s_decay[tid] = __expf(s_cs[63] - s_cs[tid]);
    a_cs[((size_t)c * NHEADS + h) * CHUNKL + tid] = s_cs[tid];
  }
  __syncthreads();

  // stage tiles (fp32 -> bf16)
  for (int e = tid; e < CHUNKL * DSTATE; e += 256) {
    const int i = e >> 7, n = e & 127;
    const float* row = xconv + (size_t)(l0 + i) * CONVDIM;
    const float bv = row[DINNER + n];
    const float cv = row[DINNER + DSTATE + n];
    sC [i * SBST + n] = (bf16_t)cv;
    sBW[i * SBST + n] = (bf16_t)bv;
    sBdT[n * SXT + i] = (bf16_t)(bv * s_decay[i]);
  }
  for (int e = tid; e < CHUNKL * HEADDIM; e += 256) {
    const int i = e >> 6, p = e & 63;
    const float xv = xconv[(size_t)(l0 + i) * CONVDIM + h * HEADDIM + p];
    sxdT[p * SXT + i] = (bf16_t)(xv * s_dt[i]);
  }
  __syncthreads();

  // scores = C @ B^T : 64x64 = 16 tiles, 2 per wave, K=128
  v8f sacc[2];
#pragma unroll
  for (int j = 0; j < 2; ++j) {
    const int t = wave * 2 + j, mt = t >> 2, nt = t & 3;
    v8f acc = {};
#pragma unroll
    for (int kk = 0; kk < 128; kk += 32) {
      v16bf a = frag_a(&sC [mt * 16 * SBST], SBST, lane, kk);
      v16bf b = frag_b(&sBW[nt * 16 * SBST], SBST, lane, kk);
      acc = wmma_bf16(a, b, acc);
    }
    sacc[j] = acc;
  }
  __syncthreads();   // done reading B tile; reuse region for W
  bf16_t* sW = sBW;
#pragma unroll
  for (int j = 0; j < 2; ++j) {
    const int t = wave * 2 + j, mt = t >> 2, nt = t & 3;
#pragma unroll
    for (int v = 0; v < 8; ++v) {
      const int lr = mt * 16 + v + 8 * fhi;
      const int sc = nt * 16 + frow;
      const float val = (lr >= sc) ? sacc[j][v] * __expf(s_cs[lr] - s_cs[sc]) : 0.f;
      sW[lr * SXT + sc] = (bf16_t)val;
    }
  }
  __syncthreads();

  // Y_diag = W @ xdt : output [l][p], B-op = xdT [p][s]
#pragma unroll
  for (int j = 0; j < 2; ++j) {
    const int t = wave * 2 + j, mt = t >> 2, pt = t & 3;
    v8f acc = {};
#pragma unroll
    for (int kk = 0; kk < 64; kk += 32) {
      v16bf a = frag_a(&sW  [mt * 16 * SXT], SXT, lane, kk);
      v16bf b = frag_b(&sxdT[pt * 16 * SXT], SXT, lane, kk);
      acc = wmma_bf16(a, b, acc);
    }
#pragma unroll
    for (int v = 0; v < 8; ++v) {
      const int lr = mt * 16 + v + 8 * fhi, p = pt * 16 + frow;
      Y[(size_t)(l0 + lr) * DINNER + h * HEADDIM + p] = acc[v];
    }
  }

  // states[p][n] = sum_l xdT[p][l] * BdT[n][l]  : 64x128 = 32 tiles, 4/wave
#pragma unroll
  for (int j = 0; j < 4; ++j) {
    const int t = wave * 4 + j, mt = t >> 3, nt = t & 7;
    v8f acc = {};
#pragma unroll
    for (int kk = 0; kk < 64; kk += 32) {
      v16bf a = frag_a(&sxdT[mt * 16 * SXT], SXT, lane, kk);
      v16bf b = frag_b(&sBdT[nt * 16 * SXT], SXT, lane, kk);
      acc = wmma_bf16(a, b, acc);
    }
#pragma unroll
    for (int v = 0; v < 8; ++v) {
      const int p = mt * 16 + v + 8 * fhi, n = nt * 16 + frow;
      states[(((size_t)c * NHEADS + h) * HEADDIM + p) * DSTATE + n] = acc[v];
    }
  }
}

// inter-chunk recurrence: NS_0 = 0; NS_{z+1} = NS_z * exp(chunksum_z) + states_z
__global__ __launch_bounds__(256) void ssd_rec_k(const float* __restrict__ states,
                                                 const float* __restrict__ a_cs,
                                                 float* __restrict__ ns) {
  const int h = blockIdx.x, tid = threadIdx.x;
  __shared__ float csum[NCHUNK];
  if (tid < NCHUNK) csum[tid] = __expf(a_cs[((size_t)tid * NHEADS + h) * CHUNKL + 63]);
  __syncthreads();
  for (int e = tid; e < HEADDIM * DSTATE; e += 256) {
    float s = 0.f;
#pragma unroll
    for (int z = 0; z < NCHUNK; ++z) {
      const size_t o = ((size_t)z * NHEADS + h) * (HEADDIM * DSTATE) + e;
      ns[o] = s;
      s = s * csum[z] + states[o];
    }
  }
}

// Y += exp(cs[l]) * (C @ NS^T) + x*D   per (chunk, head)
__global__ __launch_bounds__(256) void ssd_off_k(
    const float* __restrict__ xconv, const float* __restrict__ ns,
    const float* __restrict__ a_cs, const float* __restrict__ Dp,
    float* __restrict__ Y)
{
  __shared__ bf16_t sC [64 * SBST];   // C [l][n]  (A-op)
  __shared__ bf16_t sNS[64 * SBST];   // NS [p][n] (B-op, contraction over n)
  __shared__ float s_cs[64];
  const int tid = threadIdx.x, lane = tid & 31, wave = tid >> 5;
  const int frow = lane & 15, fhi = lane >> 4;
  const int c = blockIdx.x, h = blockIdx.y;
  const int l0 = c * CHUNKL;

  if (tid < 64) s_cs[tid] = a_cs[((size_t)c * NHEADS + h) * CHUNKL + tid];
  for (int e = tid; e < 64 * 128; e += 256) {
    const int i = e >> 7, n = e & 127;
    sC [i * SBST + n] = (bf16_t)xconv[(size_t)(l0 + i) * CONVDIM + DINNER + DSTATE + n];
    sNS[i * SBST + n] = (bf16_t)ns[(((size_t)c * NHEADS + h) * HEADDIM + i) * DSTATE + n];
  }
  __syncthreads();
  const float dskip = Dp[h];
#pragma unroll
  for (int j = 0; j < 2; ++j) {
    const int t = wave * 2 + j, mt = t >> 2, pt = t & 3;
    v8f acc = {};
#pragma unroll
    for (int kk = 0; kk < 128; kk += 32) {
      v16bf a = frag_a(&sC [mt * 16 * SBST], SBST, lane, kk);
      v16bf b = frag_b(&sNS[pt * 16 * SBST], SBST, lane, kk);
      acc = wmma_bf16(a, b, acc);
    }
#pragma unroll
    for (int v = 0; v < 8; ++v) {
      const int lr = mt * 16 + v + 8 * fhi, p = pt * 16 + frow;
      const size_t gl = (size_t)(l0 + lr);
      const size_t yo = gl * DINNER + h * HEADDIM + p;
      const float skip = xconv[gl * CONVDIM + h * HEADDIM + p] * dskip;
      Y[yo] = Y[yo] + __expf(s_cs[lr]) * acc[v] + skip;
    }
  }
}

// ======================================================================
// driver
// ======================================================================
extern "C" void kernel_launch(void* const* d_in, const int* in_sizes, int n_in,
                              void* d_out, int out_size, void* d_ws, size_t ws_size,
                              hipStream_t stream) {
  (void)in_sizes; (void)n_in; (void)out_size; (void)ws_size;
  const int*   ids       = (const int*)d_in[0];
  const float* emb       = (const float*)d_in[1];
  const float* norm_ws   = (const float*)d_in[2];
  const float* in_ws     = (const float*)d_in[3];
  const float* conv_ws   = (const float*)d_in[4];
  const float* conv_bs   = (const float*)d_in[5];
  const float* dt_biases = (const float*)d_in[6];
  const float* A_logs    = (const float*)d_in[7];
  const float* Ds        = (const float*)d_in[8];
  const float* gnorm_ws  = (const float*)d_in[9];
  const float* out_ws    = (const float*)d_in[10];
  const float* norm_f_w  = (const float*)d_in[11];
  float* logits = (float*)d_out;

  // workspace carve-up (fp32 elements), ~87 MB total
  float* ws = (float*)d_ws;
  size_t o = 0;
  float* x_resid = ws + o; o += (size_t)SEQLEN * DMODEL;            // residual stream
  float* xn      = ws + o; o += (size_t)SEQLEN * DMODEL;            // rmsnorm out
  float* zx      = ws + o; o += (size_t)SEQLEN * DINPROJ;           // in_proj out
  float* xconv   = ws + o; o += (size_t)SEQLEN * CONVDIM;           // conv+silu out
  float* dt_sp   = ws + o; o += (size_t)SEQLEN * NHEADS;
  float* dAb     = ws + o; o += (size_t)SEQLEN * NHEADS;
  float* a_cs    = ws + o; o += (size_t)NCHUNK * NHEADS * CHUNKL;
  float* Yb      = ws + o; o += (size_t)SEQLEN * DINNER;            // SSD output
  float* yn      = ws + o; o += (size_t)SEQLEN * DINNER;            // gated-norm out
  float* states  = ws + o; o += (size_t)NCHUNK * NHEADS * HEADDIM * DSTATE;
  float* nsb     = ws + o; o += (size_t)NCHUNK * NHEADS * HEADDIM * DSTATE;

  embed_k<<<(SEQLEN * DMODEL + 255) / 256, 256, 0, stream>>>(ids, emb, x_resid);

  for (int layer = 0; layer < NLAYER; ++layer) {
    rmsnorm_k<<<SEQLEN, 256, 0, stream>>>(x_resid, norm_ws + (size_t)layer * DMODEL, xn);

    gemm_nt_bf16<<<dim3((DINPROJ + GBN - 1) / GBN, SEQLEN / GBM), 256, 0, stream>>>(
        xn, in_ws + (size_t)layer * DINPROJ * DMODEL, zx, nullptr,
        SEQLEN, DINPROJ, DMODEL);

    conv_silu_k<<<(SEQLEN * CONVDIM + 255) / 256, 256, 0, stream>>>(
        zx, conv_ws + (size_t)layer * CONVDIM * DCONV,
        conv_bs + (size_t)layer * CONVDIM, xconv);

    dt_k<<<(SEQLEN * NHEADS + 255) / 256, 256, 0, stream>>>(
        zx, dt_biases + (size_t)layer * NHEADS, A_logs + (size_t)layer * NHEADS,
        dt_sp, dAb);

    ssd_intra_k<<<dim3(NCHUNK, NHEADS), 256, 0, stream>>>(
        xconv, dt_sp, dAb, a_cs, Yb, states);

    ssd_rec_k<<<NHEADS, 256, 0, stream>>>(states, a_cs, nsb);

    ssd_off_k<<<dim3(NCHUNK, NHEADS), 256, 0, stream>>>(
        xconv, nsb, a_cs, Ds + (size_t)layer * NHEADS, Yb);

    gated_rmsnorm_k<<<SEQLEN, 256, 0, stream>>>(
        Yb, zx, gnorm_ws + (size_t)layer * DINNER, yn);

    gemm_nt_bf16<<<dim3(DMODEL / GBN, SEQLEN / GBM), 256, 0, stream>>>(
        yn, out_ws + (size_t)layer * DMODEL * DINNER, x_resid, x_resid,
        SEQLEN, DMODEL, DINNER);
  }

  rmsnorm_k<<<SEQLEN, 256, 0, stream>>>(x_resid, norm_f_w, xn);

  gemm_nt_bf16<<<dim3((VOCAB + GBN - 1) / GBN, SEQLEN / GBM), 256, 0, stream>>>(
      xn, emb, logits, nullptr, SEQLEN, VOCAB, DMODEL);
}
